// DISCNet_39015482916948
// MI455X (gfx1250) — matrix operand order repaired and checked by reading, not compiled, not run on hardware
//
#include <hip/hip_runtime.h>

// ---------------------------------------------------------------------------
// DISCNet forward for gfx1250 (MI455X).
// Every 3x3 conv -> f16 im2col + f16 packed weights -> double-buffered,
// async-to-LDS staged implicit GEMM on v_wmma_f32_16x16x32_f16.
// Block = 256 thr = 8 wave32 computing a 64(M) x 32(N) macro-tile; K panels
// of 64 are DMA'd into LDS with GLOBAL_LOAD_ASYNC_TO_LDS_B128 (ASYNCcnt),
// fragments read back via ds_load_b128, 2 WMMAs per panel.
// FAC stage fully fused (1x1 kernel prediction + relu + dynamic 5x5 conv,
// replicate pad) to avoid ~740MB of kernel-tensor HBM traffic.
// ---------------------------------------------------------------------------

typedef __attribute__((ext_vector_type(16))) _Float16 v16h;
typedef __attribute__((ext_vector_type(8)))  _Float16 v8h;
typedef __attribute__((ext_vector_type(8)))  float    v8f;

static inline int cdiv_ll(long long a, long long b) { return (int)((a + b - 1) / b); }

// --------------------------- im2col (f32 -> f16) ---------------------------
// col[m][k], m = n*OH*OW + oh*OW + ow ; k = ci*9 + r*3 + s ; zero pad k >= C*9
__global__ void im2col3x3_f16(const float* __restrict__ x, _Float16* __restrict__ col,
                              int N, int C, int H, int W, int stride,
                              int OH, int OW, int Kp)
{
    long long idx = (long long)blockIdx.x * blockDim.x + threadIdx.x;
    long long total = (long long)N * OH * OW * Kp;
    if (idx >= total) return;
    int k = (int)(idx % Kp);
    long long m = idx / Kp;
    float v = 0.0f;
    int K = C * 9;
    if (k < K) {
        int ci = k / 9, rs = k % 9, r = rs / 3, s = rs % 3;
        int ow = (int)(m % OW); long long t = m / OW;
        int oh = (int)(t % OH); int n = (int)(t / OH);
        int ih = oh * stride + r - 1;
        int iw = ow * stride + s - 1;
        if (ih >= 0 && ih < H && iw >= 0 && iw < W)
            v = x[(((size_t)n * C + ci) * H + ih) * W + iw];
    }
    col[idx] = (_Float16)v;
}

// ------------------------ weight pack (OIHW -> f16) ------------------------
__global__ void pack_weights_f16(const float* __restrict__ w, _Float16* __restrict__ wf,
                                 int OC, int K, int Kp, int OCp)
{
    long long idx = (long long)blockIdx.x * blockDim.x + threadIdx.x;
    long long total = (long long)OCp * Kp;
    if (idx >= total) return;
    int k  = (int)(idx % Kp);
    int oc = (int)(idx / Kp);
    float v = (oc < OC && k < K) ? w[(size_t)oc * K + k] : 0.0f;
    wf[idx] = (_Float16)v;
}

// ----------------------- WMMA implicit-GEMM conv core ----------------------
// D[m][oc] = sum_k col[m][k] * wgt[oc][k] ; epilogue: +bias, relu?, +res?
// Macro-tile 64(M) x 32(N) per block; 8 waves as 4(M) x 2(N) 16x16 WMMA
// tiles. K in panels of 64, double-buffered in LDS via async-to-LDS DMA.
#define GBM   64
#define GBN   32
#define KSTEP 64
#define LDA   72   // LDS row stride (halves): 144B = 36 banks -> conflict-free rows
__global__ __launch_bounds__(256)
void gemm_conv_wmma(const _Float16* __restrict__ col, const _Float16* __restrict__ wgt,
                    const float* __restrict__ bias, float* __restrict__ out,
                    const float* __restrict__ res, int relu,
                    int Kp, int OC, int OCp, int HW)
{
    __shared__ _Float16 As[2][GBM * LDA];
    __shared__ _Float16 Bs[2][GBN * LDA];

    const int tid  = threadIdx.x;
    const int wave = tid >> 5, lane = tid & 31;
    const int wm = wave & 3, wn = wave >> 2;          // wave's 16x16 tile in macro-tile
    const int m0  = blockIdx.x * GBM;                 // M origin (always in range)
    const int oc0 = blockIdx.y * GBN;                 // N origin (may exceed OCp)
    const int g = lane >> 4, l16 = lane & 15;

    // Cooperative async fill coordinates.
    // A panel: 64 rows x 64 halves; each thread DMAs 2 x b128 (second at +64B
    // via INST_OFFSET, which the ISA adds to BOTH the LDS and global address).
    // B panel: 32 rows x 64 halves; each thread DMAs 1 x b128.
    const int arow = tid >> 2, aseg = (tid & 3) << 3;
    const int brow = tid >> 3, bseg = (tid & 7) << 3;
    const unsigned aGO = (unsigned)(((size_t)(m0 + arow) * Kp + aseg) * 2);
    const unsigned bGO = (unsigned)(((size_t)(oc0 + brow) * Kp + bseg) * 2);
    const bool bOk = (oc0 + brow) < OCp;   // OOB B rows never loaded; their
                                           // results are discarded in epilogue
    // LDS byte addresses (low 32 bits of generic pointer = LDS offset).
    const unsigned aL0 = (unsigned)(uintptr_t)&As[0][arow * LDA + aseg];
    const unsigned bL0 = (unsigned)(uintptr_t)&Bs[0][brow * LDA + bseg];
    const unsigned aStride = GBM * LDA * 2u;
    const unsigned bStride = GBN * LDA * 2u;

    // DMA one K panel into LDS buffer `buf` (tracked by ASYNCcnt).
    auto issue_panel = [&](int buf, int k0) {
        unsigned al = aL0 + (unsigned)buf * aStride;
        unsigned ga = aGO + (unsigned)k0 * 2u;
        asm volatile("global_load_async_to_lds_b128 %0, %1, %2"
                     :: "v"(al), "v"(ga), "s"(col) : "memory");
        asm volatile("global_load_async_to_lds_b128 %0, %1, %2 offset:64"
                     :: "v"(al), "v"(ga), "s"(col) : "memory");
        if (bOk) {
            unsigned bl = bL0 + (unsigned)buf * bStride;
            unsigned gb = bGO + (unsigned)k0 * 2u;
            asm volatile("global_load_async_to_lds_b128 %0, %1, %2"
                         :: "v"(bl), "v"(gb), "s"(wgt) : "memory");
        }
    };

    v8f acc = {};
    const int nPanels = Kp / KSTEP;        // Kp is a multiple of 64
    int buf = 0;
    issue_panel(0, 0);
    for (int p = 0; p < nPanels; ++p) {
        // Own panel DMA done, then block-wide barrier => every wave's portion
        // of LDS is visible to all waves.
        asm volatile("s_wait_asynccnt 0x0" ::: "memory");
        __syncthreads();
        // Prefetch next panel into the other buffer. Safe: the barrier above
        // proves all waves finished *reading* that buffer last iteration.
        if (p + 1 < nPanels) issue_panel(buf ^ 1, (p + 1) * KSTEP);

        // Fragment reads from LDS (ds_load_b128) + 2 WMMAs.
        const _Float16* aB = As[0] + (size_t)buf * (GBM * LDA)
                                   + (wm * 16 + l16) * LDA + (g << 3);
        const _Float16* bB = Bs[0] + (size_t)buf * (GBN * LDA)
                                   + (wn * 16 + l16) * LDA + (g << 3);
#pragma unroll
        for (int kb = 0; kb < KSTEP; kb += 32) {
            v8h a_lo = *(const v8h*)(aB + kb);        // k = kb + 8g .. +7
            v8h a_hi = *(const v8h*)(aB + kb + 16);   // k = kb + 16+8g .. +7
            v8h b_lo = *(const v8h*)(bB + kb);        // k = kb + 8g .. +7
            v8h b_hi = *(const v8h*)(bB + kb + 16);   // k = kb + 16+8g .. +7
            v16h A, B;
#pragma unroll
            for (int i = 0; i < 8; ++i) {
                A[i] = a_lo[i]; A[i + 8] = a_hi[i];
                B[i] = b_lo[i]; B[i + 8] = b_hi[i];
            }
            acc = __builtin_amdgcn_wmma_f32_16x16x32_f16(
                /*neg_a=*/false, A, /*neg_b=*/false, B,
                /*c_mod=*/(short)0, acc, /*reuse_a=*/false, /*reuse_b=*/false);
        }
        buf ^= 1;
    }

    // Epilogue. Waves whose N tile is past OCp only skip here (they still hit
    // all barriers above, so no divergent-barrier hazard).
    const int ocD = oc0 + wn * 16 + l16;
    if (ocD >= OC) return;                      // also covers OCp padding cols
    const float bv = bias[ocD];
    // D layout: lane holds (row = r + 8g, col = l16) for r = 0..7
#pragma unroll
    for (int r = 0; r < 8; ++r) {
        int mD = m0 + wm * 16 + (g << 3) + r;
        int n  = (mD >= HW) ? 1 : 0;            // batch == 2 by construction
        int hw = mD - n * HW;
        size_t oi = (size_t)(n * OC + ocD) * HW + hw;
        float v = acc[r] + bv;
        if (relu) v = fmaxf(v, 0.0f);
        if (res) v += res[oi];
        out[oi] = v;
    }
}

// ------------- fused FAC: 1x1 kernel-prediction + dynamic 5x5 conv ---------
// kernel[n,c,t,h,w] = relu( sum_ci wk[(c*25+t)*C+ci] * g[n,ci,h,w] + bk[c*25+t] )
// out[n,c,h,w]      = sum_{i,j} y_replpad[n,c,h+i-2,w+j-2] * kernel[..,i*5+j,..]
__global__ void fac_fused(const float* __restrict__ y, const float* __restrict__ g,
                          const float* __restrict__ wk, const float* __restrict__ bk,
                          float* __restrict__ out, int N, int C, int H, int W)
{
    long long idx = (long long)blockIdx.x * blockDim.x + threadIdx.x;
    long long total = (long long)N * C * H * W;
    if (idx >= total) return;
    int w = (int)(idx % W); long long t = idx / W;
    int h = (int)(t % H);   t /= H;
    int c = (int)(t % C);   int n = (int)(t / C);

    float kv[25];
#pragma unroll
    for (int i = 0; i < 25; ++i) kv[i] = bk[c * 25 + i];

    const float* wrow = wk + (size_t)(c * 25) * C;
    for (int ci = 0; ci < C; ++ci) {
        float gv = g[(((size_t)n * C + ci) * H + h) * W + w];
#pragma unroll
        for (int i = 0; i < 25; ++i)
            kv[i] = fmaf(gv, wrow[(size_t)i * C + ci], kv[i]);
    }

    const float* yc = y + (((size_t)n * C + c) * H) * W;
    float acc = 0.0f;
#pragma unroll
    for (int i = 0; i < 5; ++i) {
        int ih = h + i - 2; ih = ih < 0 ? 0 : (ih >= H ? H - 1 : ih);
#pragma unroll
        for (int j = 0; j < 5; ++j) {
            int iw = w + j - 2; iw = iw < 0 ? 0 : (iw >= W ? W - 1 : iw);
            float kvv = fmaxf(kv[i * 5 + j], 0.0f);
            acc = fmaf(kvv, yc[(size_t)ih * W + iw], acc);
        }
    }
    out[idx] = acc;
}

// --------------- pixel-shuffle(2) + ReLU + skip-add (decoder) --------------
// in: (N,4C,H,W) -> out: (N,C,2H,2W) = relu(shuffle(in)) + skip
__global__ void pixshuf_relu_add(const float* __restrict__ in, const float* __restrict__ skip,
                                 float* __restrict__ out, int N, int C, int H, int W)
{
    long long idx = (long long)blockIdx.x * blockDim.x + threadIdx.x;
    long long total = (long long)N * C * 4 * H * W;
    if (idx >= total) return;
    int OW = 2 * W, OH = 2 * H;
    int ow = (int)(idx % OW); long long t = idx / OW;
    int oh = (int)(t % OH);   t /= OH;
    int c  = (int)(t % C);    int n = (int)(t / C);
    int a = oh & 1, b = ow & 1, h = oh >> 1, w = ow >> 1;
    float v = in[((((size_t)n * (4 * C)) + (c * 4 + a * 2 + b)) * H + h) * W + w];
    v = fmaxf(v, 0.0f);
    out[idx] = v + skip[idx];
}

// ---------------------------------------------------------------------------
extern "C" void kernel_launch(void* const* d_in, const int* in_sizes, int n_in,
                              void* d_out, int out_size, void* d_ws, size_t ws_size,
                              hipStream_t stream)
{
    (void)in_sizes; (void)out_size; (void)ws_size;
    const int NB = 2;  // batch (fixed by reference)

    const float* x    = (const float*)d_in[0];
    const float* cond = (const float*)d_in[1];

    // params: (w,b) pairs in dict-insertion order
    const float* p_w[64];
    const float* p_b[64];
    int np = 0;
    for (int i = 2; i + 1 < n_in; i += 2) { p_w[np] = (const float*)d_in[i]; p_b[np] = (const float*)d_in[i + 1]; ++np; }
    // pair map: enc0..2 = 0..14 ; cond_enc0..2 = 15..29 ; dk0 = 30..35 (k=35),
    // dk1 = 36..41, dk2 = 42..47 ; dec1_up.c = 48 ; dec1 rbs = 49..52 ;
    // dec2_up.c = 53 ; dec2 rbs = 54..57 ; out.c = 58

    // --- workspace carving ---
    char* ws = (char*)d_ws;
    size_t off = 0;
    auto alloc = [&](size_t bytes) -> char* {
        char* p = ws + off;
        off += (bytes + 255) & ~(size_t)255;
        return p;
    };
    const size_t S0 = (size_t)NB * 32 * 256 * 256;           // largest feature map (elems)
    float* Fa    = (float*)alloc(S0 * 4);
    float* Fb    = (float*)alloc(S0 * 4);
    float* Ya    = (float*)alloc(S0 * 4);
    float* Yb    = (float*)alloc(S0 * 4);
    float* G     = (float*)alloc(S0 * 4);
    float* t1    = (float*)alloc(S0 * 4);
    float* t2    = (float*)alloc(S0 * 4);
    float* outs0 = (float*)alloc(S0 * 4);
    float* outs1 = (float*)alloc((size_t)NB * 64 * 128 * 128 * 4);
    float* outs2 = (float*)alloc((size_t)NB * 128 * 64 * 64 * 4);
    _Float16* colb = (_Float16*)alloc((size_t)NB * 256 * 256 * 320 * 2); // max M*Kp64 (f16)
    _Float16* wfb  = (_Float16*)alloc((size_t)256 * 1152 * 2);           // max OCp*Kp64 (f16)

    // --- one 3x3 conv: im2col -> pack weights -> WMMA GEMM (+bias/relu/res) ---
    auto conv = [&](const float* in, int Cin, int H, int W, int stride, int p,
                    int OC, bool relu, const float* res, float* out) {
        int OH = (H - 1) / stride + 1;
        int OW = (W - 1) / stride + 1;
        int K  = Cin * 9;
        int Kp = (K + 63) & ~63;                 // multiple of KSTEP (64)
        int OCp = (OC + 15) & ~15;
        long long Mtot = (long long)NB * OH * OW;            // multiple of 64 for our shapes
        long long colTot = Mtot * Kp;
        im2col3x3_f16<<<cdiv_ll(colTot, 256), 256, 0, stream>>>(
            in, colb, NB, Cin, H, W, stride, OH, OW, Kp);
        pack_weights_f16<<<cdiv_ll((long long)OCp * Kp, 256), 256, 0, stream>>>(
            p_w[p], wfb, OC, K, Kp, OCp);
        dim3 grid((unsigned)(Mtot / GBM), (unsigned)((OCp + GBN - 1) / GBN));
        gemm_conv_wmma<<<grid, 256, 0, stream>>>(
            colb, wfb, p_b[p], out, res, relu ? 1 : 0,
            Kp, OC, OCp, OH * OW);
    };

    // --- encoder / dk block: conv(+relu) then two residual blocks ---
    auto enc_block = [&](const float* in, int Cin, int H, int W, int stride,
                         int p0, int OC, float* out) {
        int OH = (H - 1) / stride + 1;
        int OW = (W - 1) / stride + 1;
        conv(in, Cin, H, W, stride, p0, OC, true, nullptr, t1);
        conv(t1, OC, OH, OW, 1, p0 + 1, OC, true,  nullptr, t2);   // rb0.c1
        conv(t2, OC, OH, OW, 1, p0 + 2, OC, false, t1, t1);        // rb0.c2 + skip
        conv(t1, OC, OH, OW, 1, p0 + 3, OC, true,  nullptr, t2);   // rb1.c1
        conv(t2, OC, OH, OW, 1, p0 + 4, OC, false, t1, out);       // rb1.c2 + skip
    };

    auto fac = [&](const float* y, const float* g, int pk, float* out,
                   int C, int H, int W) {
        long long total = (long long)NB * C * H * W;
        fac_fused<<<cdiv_ll(total, 256), 256, 0, stream>>>(
            y, g, p_w[pk], p_b[pk], out, NB, C, H, W);
    };

    // ================= scale 0 (32ch @ 256x256) =================
    enc_block(cond, 3, 256, 256, 1, 15, 32, Fa);    // cond_enc0
    enc_block(Fa,  32, 256, 256, 1, 30, 32, G);     // dk0 features
    enc_block(x,    3, 256, 256, 1,  0, 32, Ya);    // enc0
    fac(Ya, G, 35, outs0, 32, 256, 256);

    // ================= scale 1 (64ch @ 128x128) =================
    enc_block(Fa, 32, 256, 256, 2, 20, 64, Fb);     // cond_enc1
    enc_block(Fb, 64, 128, 128, 1, 36, 64, G);      // dk1 features
    enc_block(Ya, 32, 256, 256, 2,  5, 64, Yb);     // enc1
    fac(Yb, G, 41, outs1, 64, 128, 128);

    // ================= scale 2 (128ch @ 64x64) ==================
    enc_block(Fb,  64, 128, 128, 2, 25, 128, Fa);   // cond_enc2
    enc_block(Fa, 128,  64,  64, 1, 42, 128, G);    // dk2 features
    enc_block(Yb,  64, 128, 128, 2, 10, 128, Ya);   // enc2
    fac(Ya, G, 47, outs2, 128, 64, 64);

    // ================= decoder ==================================
    // dec1: conv 128->256 @64x64, shuffle -> 64 @128x128, relu, + outs1
    conv(outs2, 128, 64, 64, 1, 48, 256, false, nullptr, t1);
    {
        long long tot = (long long)NB * 64 * 4 * 64 * 64;
        pixshuf_relu_add<<<cdiv_ll(tot, 256), 256, 0, stream>>>(t1, outs1, Yb, NB, 64, 64, 64);
    }
    conv(Yb, 64, 128, 128, 1, 49, 64, true,  nullptr, t1);
    conv(t1, 64, 128, 128, 1, 50, 64, false, Yb, Yb);
    conv(Yb, 64, 128, 128, 1, 51, 64, true,  nullptr, t1);
    conv(t1, 64, 128, 128, 1, 52, 64, false, Yb, Yb);

    // dec2: conv 64->128 @128x128, shuffle -> 32 @256x256, relu, + outs0
    conv(Yb, 64, 128, 128, 1, 53, 128, false, nullptr, t1);
    {
        long long tot = (long long)NB * 32 * 4 * 128 * 128;
        pixshuf_relu_add<<<cdiv_ll(tot, 256), 256, 0, stream>>>(t1, outs0, Fa, NB, 32, 128, 128);
    }
    conv(Fa, 32, 256, 256, 1, 54, 32, true,  nullptr, t1);
    conv(t1, 32, 256, 256, 1, 55, 32, false, Fa, Fa);
    conv(Fa, 32, 256, 256, 1, 56, 32, true,  nullptr, t1);
    conv(t1, 32, 256, 256, 1, 57, 32, false, Fa, Fa);

    // out: relu(conv 32->3) + x  (OC padded to 16 inside gemm, stores guarded)
    conv(Fa, 32, 256, 256, 1, 58, 3, true, x, (float*)d_out);
}
